// PPO_GNN_10703058502077
// MI455X (gfx1250) — compile-verified, matrix operand order checked by reading, not run on hardware
//
#include <hip/hip_runtime.h>
#include <hip/hip_bf16.h>

typedef __bf16 bf16;
typedef __attribute__((ext_vector_type(16))) __bf16 bf16x16;
typedef __attribute__((ext_vector_type(8)))  float  f32x8;

// ---------------------------------------------------------------------------
// Kernel 1: normalized adjacency (adj + I) * deg^-1, cast to bf16.
// One 256-thread block per row (1024 cols, 4 per thread).
// ---------------------------------------------------------------------------
__global__ __launch_bounds__(256)
void normalize_adj(const float* __restrict__ adj, bf16* __restrict__ out) {
  const int row_id = blockIdx.x;            // 0..32767  (b*1024 + n)
  const int n = row_id & 1023;
  const float* arow = adj + (long long)row_id * 1024;
  __shared__ float red[256];

  float vals[4];
  float s = 0.f;
#pragma unroll
  for (int i = 0; i < 4; ++i) {
    int m = threadIdx.x + 256 * i;
    float v = arow[m] + (m == n ? 1.f : 0.f);
    vals[i] = v;
    s += v;
  }
  red[threadIdx.x] = s;
  __syncthreads();
  for (int off = 128; off > 0; off >>= 1) {
    if ((int)threadIdx.x < off) red[threadIdx.x] += red[threadIdx.x + off];
    __syncthreads();
  }
  const float deg = red[0];
  const float inv = (deg == 0.f) ? 0.f : 1.f / deg;

  bf16* orow = out + (long long)row_id * 1024;
#pragma unroll
  for (int i = 0; i < 4; ++i) {
    int m = threadIdx.x + 256 * i;
    orow[m] = (bf16)(vals[i] * inv);
  }
}

// ---------------------------------------------------------------------------
// Kernel 2: node features [32,1024,32] f32 -> transposed bf16 [32,32,1024]
// ---------------------------------------------------------------------------
__global__ __launch_bounds__(256)
void transpose_x0(const float* __restrict__ x, bf16* __restrict__ xT) {
  int idx = blockIdx.x * 256 + threadIdx.x;      // 1,048,576 total
  int f = idx & 31;
  int n = (idx >> 5) & 1023;
  int b = idx >> 15;
  xT[((long long)b * 32 + f) * 1024 + n] = (bf16)x[idx];
}

// ---------------------------------------------------------------------------
// Kernel 3: weight prep: f32 [K,Nin] -> bf16 transposed [Npad,K], zero-padded
// ---------------------------------------------------------------------------
__global__ __launch_bounds__(256)
void prep_weight(const float* __restrict__ W, bf16* __restrict__ WT,
                 int K, int Nin, int Npad) {
  int idx = blockIdx.x * 256 + threadIdx.x;
  if (idx >= K * Npad) return;
  int k = idx % K;
  int n = idx / K;
  WT[(long long)n * K + k] = (bf16)((n < Nin) ? W[(long long)k * Nin + n] : 0.f);
}

__global__ void pad_bias16(const float* __restrict__ b, float* __restrict__ out, int len) {
  int i = threadIdx.x;
  if (i < 16) out[i] = (i < len) ? b[i] : 0.f;
}

// ---------------------------------------------------------------------------
// Generic WMMA GEMM: D[M,16*NT] = A[M,K](bf16,row-major) x B(bf16,[N,K] i.e.
// pre-transposed) (+bias, relu). One wave owns a 16-row strip and all NT
// column tiles; K-loop of v_wmma_f32_16x16x32_bf16.
//   OUT_MODE 0: f32 row-major (first out_cols columns)
//   OUT_MODE 1: bf16 row-major
//   OUT_MODE 2: bf16 transposed per 1024-row batch: O[b][col][token]
// ---------------------------------------------------------------------------
template <int NT, int OUT_MODE, bool RELU>
__global__ __launch_bounds__(256)
void wmma_gemm(const bf16* __restrict__ A, long long strideA,
               const bf16* __restrict__ Bm, long long strideB,
               const float* __restrict__ bias,
               int rows_per_batch, int K,
               void* __restrict__ Out, long long strideOut,
               int ld_out, int out_cols) {
  const int wave = (int)((blockIdx.x * blockDim.x + threadIdx.x) >> 5);
  const int lane = threadIdx.x & 31;
  const int ll = lane & 15;     // row (A) / col (B) within tile
  const int lh = lane >> 4;     // K half select
  const int row0 = wave * 16;
  const int batch = row0 / rows_per_batch;
  const int row_in_b = row0 % rows_per_batch;

  const bf16* Ab = A + (long long)batch * strideA + (long long)(row_in_b + ll) * K + lh * 16;
  const bf16* Bb = Bm + (long long)batch * strideB + (long long)ll * K + lh * 16;

  f32x8 acc[NT];
#pragma unroll
  for (int t = 0; t < NT; ++t) {
    f32x8 z = {0.f, 0.f, 0.f, 0.f, 0.f, 0.f, 0.f, 0.f};
    acc[t] = z;
  }

  for (int k = 0; k < K; k += 32) {
    bf16x16 afrag = *(const bf16x16*)(Ab + k);
#pragma unroll
    for (int t = 0; t < NT; ++t) {
      bf16x16 bfrag = *(const bf16x16*)(Bb + (long long)t * 16 * K + k);
      acc[t] = __builtin_amdgcn_wmma_f32_16x16x32_bf16(
          false, afrag, false, bfrag, (short)0, acc[t], false, false);
    }
  }

#pragma unroll
  for (int t = 0; t < NT; ++t) {
    const int col = t * 16 + ll;
    const float bv = bias ? bias[col] : 0.f;
    if (OUT_MODE == 0) {
      float* O = (float*)Out;
      if (col < out_cols) {
#pragma unroll
        for (int v = 0; v < 8; ++v) {
          float x = acc[t][v] + bv;
          if (RELU) x = fmaxf(x, 0.f);
          O[(long long)(row0 + lh * 8 + v) * ld_out + col] = x;
        }
      }
    } else if (OUT_MODE == 1) {
      bf16* O = (bf16*)Out;
#pragma unroll
      for (int v = 0; v < 8; ++v) {
        float x = acc[t][v] + bv;
        if (RELU) x = fmaxf(x, 0.f);
        O[(long long)(row0 + lh * 8 + v) * ld_out + col] = (bf16)x;
      }
    } else {
      bf16* O = (bf16*)Out + (long long)(row0 >> 10) * strideOut;
      const int tr = (row0 & 1023) + lh * 8;
#pragma unroll
      for (int v = 0; v < 8; ++v) {
        float x = acc[t][v] + bv;
        if (RELU) x = fmaxf(x, 0.f);
        O[(long long)col * 1024 + tr + v] = (bf16)x;   // 8 contiguous bf16/lane
      }
    }
  }
}

// ---------------------------------------------------------------------------
extern "C" void kernel_launch(void* const* d_in, const int* in_sizes, int n_in,
                              void* d_out, int out_size, void* d_ws, size_t ws_size,
                              hipStream_t stream) {
  (void)in_sizes; (void)n_in; (void)out_size; (void)ws_size;

  const float* node   = (const float*)d_in[0];   // [32,1024,32]
  const float* adj    = (const float*)d_in[1];   // [32,1024,1024]
  const float* gcn_w1 = (const float*)d_in[2];
  const float* gcn_b1 = (const float*)d_in[3];
  const float* gcn_w2 = (const float*)d_in[4];
  const float* gcn_b2 = (const float*)d_in[5];
  const float* pol_w1 = (const float*)d_in[6];
  const float* pol_b1 = (const float*)d_in[7];
  const float* pol_w2 = (const float*)d_in[8];
  const float* pol_b2 = (const float*)d_in[9];
  const float* pol_w3 = (const float*)d_in[10];
  const float* pol_b3 = (const float*)d_in[11];
  const float* val_w1 = (const float*)d_in[12];
  const float* val_b1 = (const float*)d_in[13];
  const float* val_w2 = (const float*)d_in[14];
  const float* val_b2 = (const float*)d_in[15];
  const float* val_w3 = (const float*)d_in[16];
  const float* val_b3 = (const float*)d_in[17];

  // ---- workspace carving (bf16 unless noted) ----
  char* p = (char*)d_ws;
  auto take = [&](size_t elems, size_t esz) { char* r = p; p += elems * esz; return r; };
  bf16* adjN = (bf16*)take(33554432ull, 2);  // [32,1024,1024]
  bf16* x0T  = (bf16*)take(1048576ull, 2);   // [32,32,1024]
  bf16* msg1 = (bf16*)take(1048576ull, 2);   // [32768,32]
  bf16* x1T  = (bf16*)take(2097152ull, 2);   // [32,64,1024]
  bf16* msg2 = (bf16*)take(2097152ull, 2);   // [32768,64]
  bf16* x2   = (bf16*)take(2097152ull, 2);   // [32768,64]
  bf16* hp1  = (bf16*)take(4194304ull, 2);   // [32768,128]
  bf16* hp2  = (bf16*)take(4194304ull, 2);
  bf16* hv1  = (bf16*)take(4194304ull, 2);
  bf16* hv2  = (bf16*)take(4194304ull, 2);
  bf16* w1T  = (bf16*)take(64 * 32, 2);
  bf16* w2T  = (bf16*)take(64 * 64, 2);
  bf16* pw1T = (bf16*)take(128 * 64, 2);
  bf16* pw2T = (bf16*)take(128 * 128, 2);
  bf16* pw3T = (bf16*)take(16 * 128, 2);
  bf16* vw1T = (bf16*)take(128 * 64, 2);
  bf16* vw2T = (bf16*)take(128 * 128, 2);
  bf16* vw3T = (bf16*)take(16 * 128, 2);
  float* pb3 = (float*)take(16, 4);
  float* vb3 = (float*)take(16, 4);

  // ---- prep ----
  normalize_adj<<<32768, 256, 0, stream>>>(adj, adjN);
  transpose_x0<<<4096, 256, 0, stream>>>(node, x0T);
  prep_weight<<<8, 256, 0, stream>>>(gcn_w1, w1T, 32, 64, 64);
  prep_weight<<<16, 256, 0, stream>>>(gcn_w2, w2T, 64, 64, 64);
  prep_weight<<<32, 256, 0, stream>>>(pol_w1, pw1T, 64, 128, 128);
  prep_weight<<<64, 256, 0, stream>>>(pol_w2, pw2T, 128, 128, 128);
  prep_weight<<<8, 256, 0, stream>>>(pol_w3, pw3T, 128, 8, 16);
  prep_weight<<<32, 256, 0, stream>>>(val_w1, vw1T, 64, 128, 128);
  prep_weight<<<64, 256, 0, stream>>>(val_w2, vw2T, 128, 128, 128);
  prep_weight<<<8, 256, 0, stream>>>(val_w3, vw3T, 128, 1, 16);
  pad_bias16<<<1, 16, 0, stream>>>(pol_b3, pb3, 8);
  pad_bias16<<<1, 16, 0, stream>>>(val_b3, vb3, 1);

  const int GB = 256, TB = 256;  // 2048 waves = 32768 rows / 16 for every GEMM

  // GCN layer 1: msg1 = adjN @ x0  (per batch), then x1 = relu(msg1@w1+b1) -> transposed
  wmma_gemm<2, 1, false><<<GB, TB, 0, stream>>>(adjN, 1048576ll, x0T, 32768ll,
      nullptr, 1024, 1024, msg1, 0, 32, 32);
  wmma_gemm<4, 2, true><<<GB, TB, 0, stream>>>(msg1, 0, w1T, 0,
      gcn_b1, 32768, 32, x1T, 65536ll, 0, 64);

  // GCN layer 2: msg2 = adjN @ x1, then x2 = relu(msg2@w2+b2)
  wmma_gemm<4, 1, false><<<GB, TB, 0, stream>>>(adjN, 1048576ll, x1T, 65536ll,
      nullptr, 1024, 1024, msg2, 0, 64, 64);
  wmma_gemm<4, 1, true><<<GB, TB, 0, stream>>>(msg2, 0, w2T, 0,
      gcn_b2, 32768, 64, x2, 0, 64, 64);

  // policy head
  wmma_gemm<8, 1, true><<<GB, TB, 0, stream>>>(x2, 0, pw1T, 0,
      pol_b1, 32768, 64, hp1, 0, 128, 128);
  wmma_gemm<8, 1, true><<<GB, TB, 0, stream>>>(hp1, 0, pw2T, 0,
      pol_b2, 32768, 128, hp2, 0, 128, 128);
  wmma_gemm<1, 0, false><<<GB, TB, 0, stream>>>(hp2, 0, pw3T, 0,
      pb3, 32768, 128, (float*)d_out, 0, 8, 8);

  // value head
  wmma_gemm<8, 1, true><<<GB, TB, 0, stream>>>(x2, 0, vw1T, 0,
      val_b1, 32768, 64, hv1, 0, 128, 128);
  wmma_gemm<8, 1, true><<<GB, TB, 0, stream>>>(hv1, 0, vw2T, 0,
      val_b2, 32768, 128, hv2, 0, 128, 128);
  wmma_gemm<1, 0, false><<<GB, TB, 0, stream>>>(hv2, 0, vw3T, 0,
      vb3, 32768, 128, (float*)d_out + 262144, 0, 1, 1);
}
